// AngularAggLayer_76579266888068
// MI455X (gfx1250) — matrix operand order repaired and checked by reading, not compiled, not run on hardware
//
#include <hip/hip_runtime.h>
#include <hip/hip_bf16.h>

#define NN 8192
#define DD 128
#define CC 10
#define TT 1024

typedef __attribute__((ext_vector_type(8)))  float  v8f;
typedef __attribute__((ext_vector_type(16))) __bf16 v16bf;
typedef __attribute__((ext_vector_type(8)))  __bf16 v8bf;

union FragB { v16bf v; v8bf h[2]; };

#define BM 128
#define BN 128
#define BK 32
#define LDS_STRIDE 40   // bf16 elements per LDS row (32 + 8 pad, keeps 16B alignment)

// ---------------------------------------------------------------------------
// Kernel: build combined complex-as-real weight (bf16, [n][k] layout) + bias
// Wbig[n][k]: n<128 (row j=n of yr):  k<128 -> Wr[j][k], k>=128 -> -Wi[j][k-128]
//             n>=128 (row j of yi):   k<128 -> Wi[j][k], k>=128 ->  Wr[j][k-128]
// bias2[n] = n<128 ? b_r[n]-b_i[n] : b_r[n-128]+b_i[n-128]
// ---------------------------------------------------------------------------
__global__ __launch_bounds__(256) void wprep_kernel(
    const float* __restrict__ Wr, const float* __restrict__ Wi,
    const float* __restrict__ br, const float* __restrict__ bi,
    __bf16* __restrict__ Wbig, float* __restrict__ bias2)
{
    int n = blockIdx.x;      // 0..255
    int k = threadIdx.x;     // 0..255
    float v;
    if (n < 128) v = (k < 128) ? Wr[n * 128 + k] : -Wi[n * 128 + (k - 128)];
    else {
        int j = n - 128;
        v = (k < 128) ? Wi[j * 128 + k] : Wr[j * 128 + (k - 128)];
    }
    Wbig[n * 256 + k] = (__bf16)v;
    if (k == 0)
        bias2[n] = (n < 128) ? (br[n] - bi[n]) : (br[n - 128] + bi[n - 128]);
}

// ---------------------------------------------------------------------------
// Kernel: per-node class projection, softmax, in_r/in_i, s_r/s_i -> S^T (bf16)
// One wave32 per node; 8 nodes per 256-thread block.
// ---------------------------------------------------------------------------
__global__ __launch_bounds__(256) void node_kernel(
    const float* __restrict__ xr, const float* __restrict__ xi,
    const float* __restrict__ Wcr, const float* __restrict__ bcr,
    const float* __restrict__ Wci, const float* __restrict__ bci,
    const float* __restrict__ thr, const float* __restrict__ thi,
    __bf16* __restrict__ St,       // [256][8192] bf16
    float* __restrict__ pnOut,     // [N][10]
    float* __restrict__ inR, float* __restrict__ inI)
{
    __shared__ __bf16 sTile[256 * 8];   // [d(0..255)][nodeLocal(0..7)]

    const int lane = threadIdx.x & 31;
    const int wave = threadIdx.x >> 5;
    const int node = blockIdx.x * 8 + wave;

    float xrv[4], xiv[4];
#pragma unroll
    for (int j = 0; j < 4; ++j) {
        int d = lane + 32 * j;
        xrv[j] = xr[(size_t)node * DD + d];
        xiv[j] = xi[(size_t)node * DD + d];
    }

    float pn[CC];
#pragma unroll
    for (int c = 0; c < CC; ++c) {
        float p0 = 0.f, p1 = 0.f, p2 = 0.f, p3 = 0.f;
#pragma unroll
        for (int j = 0; j < 4; ++j) {
            int d = lane + 32 * j;
            float wr = Wcr[c * DD + d];
            float wi = Wci[c * DD + d];
            p0 += xrv[j] * wr;   // xr . Wcr
            p1 += xiv[j] * wi;   // xi . Wci
            p2 += xiv[j] * wr;   // xi . Wcr
            p3 += xrv[j] * wi;   // xr . Wci
        }
#pragma unroll
        for (int off = 16; off > 0; off >>= 1) {
            p0 += __shfl_xor(p0, off, 32);
            p1 += __shfl_xor(p1, off, 32);
            p2 += __shfl_xor(p2, off, 32);
            p3 += __shfl_xor(p3, off, 32);
        }
        float cr = (p0 + bcr[c]) - (p1 + bci[c]);
        float ci = (p2 + bcr[c]) + (p3 + bci[c]);
        pn[c] = sqrtf(cr * cr + ci * ci);
    }

    // softmax over C, then in = sim @ theta
    float mx = pn[0];
#pragma unroll
    for (int c = 1; c < CC; ++c) mx = fmaxf(mx, pn[c]);
    float se = 0.f;
    float ex[CC];
#pragma unroll
    for (int c = 0; c < CC; ++c) { ex[c] = __expf(pn[c] - mx); se += ex[c]; }
    float inv = 1.f / se;
    float ir = 0.f, ii = 0.f;
#pragma unroll
    for (int c = 0; c < CC; ++c) {
        float sim = ex[c] * inv;
        ir += sim * thr[c];
        ii += sim * thi[c];
    }

    if (lane < CC) pnOut[(size_t)node * CC + lane] = pn[lane];
    if (lane == 0) { inR[node] = ir; inI[node] = ii; }

    // s_r = in_r*xr + in_i*xi ; s_i = in_r*xi - in_i*xr   (out = conj(in))
#pragma unroll
    for (int j = 0; j < 4; ++j) {
        int d = lane + 32 * j;
        float sr = ir * xrv[j] + ii * xiv[j];
        float si = ir * xiv[j] - ii * xrv[j];
        sTile[d * 8 + wave]         = (__bf16)sr;
        sTile[(128 + d) * 8 + wave] = (__bf16)si;
    }
    __syncthreads();
    // coalesced transpose-out: thread t writes 8 consecutive nodes of row t
    int t = threadIdx.x;
    *(uint4*)&St[(size_t)t * NN + (size_t)blockIdx.x * 8] = *(const uint4*)&sTile[t * 8];
}

// ---------------------------------------------------------------------------
// Kernel: separate_loss + supervised_loss (single block)
// ---------------------------------------------------------------------------
__global__ __launch_bounds__(256) void loss_kernel(
    const float* __restrict__ pn, const int* __restrict__ labels,
    const int* __restrict__ tidx, const float* __restrict__ thi,
    float* __restrict__ outSep, float* __restrict__ outSup)
{
    __shared__ float red[256];
    int t = threadIdx.x;
    float local = 0.f;
    for (int s = t; s < TT; s += 256) {
        int idx = tidx[s];
        int lab = labels[idx];
        const float* l = pn + (size_t)idx * CC;
        float mx = l[0];
#pragma unroll
        for (int c = 1; c < CC; ++c) mx = fmaxf(mx, l[c]);
        float se = 0.f;
#pragma unroll
        for (int c = 0; c < CC; ++c) se += __expf(l[c] - mx);
        float lse = logf(se) + mx;
        local += (l[lab] - lse);   // logp at label
    }
    red[t] = local;
    __syncthreads();
    for (int off = 128; off > 0; off >>= 1) {
        if (t < off) red[t] += red[t + off];
        __syncthreads();
    }
    if (t == 0) {
        *outSup = -(red[0] / (float)TT) * 0.1f;

        float st[CC];
#pragma unroll
        for (int c = 0; c < CC; ++c) st[c] = thi[c];
        // insertion sort ascending
        for (int a = 1; a < CC; ++a) {
            float key = st[a];
            int b = a - 1;
            while (b >= 0 && st[b] > key) { st[b + 1] = st[b]; --b; }
            st[b + 1] = key;
        }
        float sum = 0.f, mxd = 0.f;
        for (int k = 0; k < CC - 1; ++k) {
            float d = fabsf(st[k] - st[k + 1]);
            sum += d;
            mxd = fmaxf(mxd, d);
        }
        float diff = sum / (mxd + 1e-6f);
        *outSep = 0.1f / (1e-6f + diff);
    }
}

// ---------------------------------------------------------------------------
// WMMA GEMM: C[M x 256] = A(f32, MxK) @ B  with B given as Bt[n][k] bf16.
// A is converted f32->bf16 while staging to LDS. fp32 accumulation.
// EPI=0: plain store. EPI=1: + bias[col] + residual add, uniform half select.
// Block = 256 threads = 8 waves; tile 128x128; BK=32; double-buffered LDS.
// ---------------------------------------------------------------------------
template <int EPI>
__global__ __launch_bounds__(256) void gemm_wmma_kernel(
    const float*  __restrict__ A,  int lda,   // M x K fp32
    const __bf16* __restrict__ Bt, int ldb,   // 256 x K bf16, row n holds K
    int K,
    float* __restrict__ outLo, float* __restrict__ outHi, int ldOut,
    const float* __restrict__ bias,           // [256] (EPI==1)
    const float* __restrict__ addLo, const float* __restrict__ addHi)
{
    __shared__ __bf16 As[2][BM * LDS_STRIDE];
    __shared__ __bf16 Bs[2][BN * LDS_STRIDE];

    const int tid  = threadIdx.x;
    const int lane = tid & 31;
    const int wave = tid >> 5;
    const int wm   = wave & 3;   // 4 m-groups of 32 rows
    const int wn   = wave >> 2;  // 2 n-groups of 64 cols

    const int rowBlk = blockIdx.x * BM;
    const int colBlk = blockIdx.y * BN;

    const int ldr = tid >> 1;         // row within tile (0..127)
    const int ldk = (tid & 1) * 16;   // 0 or 16 (k offset)

    const float*  aSrc = A  + (size_t)(rowBlk + ldr) * lda + ldk;
    const __bf16* bSrc = Bt + (size_t)(colBlk + ldr) * ldb + ldk;

    // prefetch staging kept in scalar locals (no arrays -> no scratch spills)
    float4 pa0, pa1, pa2, pa3;
    uint4  pb0, pb1;

    auto fetch = [&](int kOff) {
        const float* ap = aSrc + kOff;
        pa0 = *(const float4*)(ap + 0);
        pa1 = *(const float4*)(ap + 4);
        pa2 = *(const float4*)(ap + 8);
        pa3 = *(const float4*)(ap + 12);
        const __bf16* bp = bSrc + kOff;
        pb0 = *(const uint4*)(bp + 0);
        pb1 = *(const uint4*)(bp + 8);
    };
    auto stage = [&](int s) {
        v8bf t0, t1;
        t0[0] = (__bf16)pa0.x; t0[1] = (__bf16)pa0.y;
        t0[2] = (__bf16)pa0.z; t0[3] = (__bf16)pa0.w;
        t0[4] = (__bf16)pa1.x; t0[5] = (__bf16)pa1.y;
        t0[6] = (__bf16)pa1.z; t0[7] = (__bf16)pa1.w;
        t1[0] = (__bf16)pa2.x; t1[1] = (__bf16)pa2.y;
        t1[2] = (__bf16)pa2.z; t1[3] = (__bf16)pa2.w;
        t1[4] = (__bf16)pa3.x; t1[5] = (__bf16)pa3.y;
        t1[6] = (__bf16)pa3.z; t1[7] = (__bf16)pa3.w;
        __bf16* Ad = As[s];
        __bf16* Bd = Bs[s];
        *(v8bf*)&Ad[ldr * LDS_STRIDE + ldk]      = t0;
        *(v8bf*)&Ad[ldr * LDS_STRIDE + ldk + 8]  = t1;
        *(uint4*)&Bd[ldr * LDS_STRIDE + ldk]     = pb0;
        *(uint4*)&Bd[ldr * LDS_STRIDE + ldk + 8] = pb1;
    };

    v8f acc[2][4];
#pragma unroll
    for (int mb = 0; mb < 2; ++mb)
#pragma unroll
        for (int nb = 0; nb < 4; ++nb)
#pragma unroll
            for (int v = 0; v < 8; ++v) acc[mb][nb][v] = 0.f;

    fetch(0);
    stage(0);

    const int KT = K / BK;
    int cur = 0;
    for (int kt = 0; kt < KT; ++kt) {
        __syncthreads();
        const bool pf = (kt + 1) < KT;
        if (pf) {
            __builtin_prefetch(aSrc + (size_t)(kt + 1) * BK, 0, 0);
            fetch((kt + 1) * BK);
        }

        const __bf16* Ab = As[cur];
        const __bf16* Bb = Bs[cur];

        // A fragments: ISA 16-bit A layout (interleaved K chunks per half-wave)
        FragB af[2];
#pragma unroll
        for (int mb = 0; mb < 2; ++mb) {
            int r  = wm * 32 + mb * 16 + (lane & 15);
            int c0 = (lane < 16) ? 0  : 8;
            int c1 = (lane < 16) ? 16 : 24;
            af[mb].h[0] = *(const v8bf*)&Ab[r * LDS_STRIDE + c0];
            af[mb].h[1] = *(const v8bf*)&Ab[r * LDS_STRIDE + c1];
        }
        // B fragments: lanes 0-15 hold K 0..7,16..23, lanes 16-31 hold K 8..15,24..31
        FragB bf4[4];
#pragma unroll
        for (int nb = 0; nb < 4; ++nb) {
            int n  = wn * 64 + nb * 16 + (lane & 15);
            int kk = (lane < 16) ? 0 : 8;
            bf4[nb].h[0] = *(const v8bf*)&Bb[n * LDS_STRIDE + kk];
            bf4[nb].h[1] = *(const v8bf*)&Bb[n * LDS_STRIDE + kk + 16];
        }

#pragma unroll
        for (int mb = 0; mb < 2; ++mb)
#pragma unroll
            for (int nb = 0; nb < 4; ++nb)
                acc[mb][nb] = __builtin_amdgcn_wmma_f32_16x16x32_bf16(
                    false, af[mb].v, false, bf4[nb].v,
                    (short)0, acc[mb][nb], false, false);

        if (pf) stage(cur ^ 1);
        cur ^= 1;
    }

    // Epilogue. Half selection is uniform per block (colBlk = 0 or 128).
    float*       outP = (colBlk == 0) ? outLo : outHi;
    const float* addP = (colBlk == 0) ? addLo : addHi;

    // C/D f32 layout — VGPR v: lanes 0-15 -> M=v, N=lane; lanes 16-31 -> M=8+v
#pragma unroll
    for (int mb = 0; mb < 2; ++mb) {
#pragma unroll
        for (int nb = 0; nb < 4; ++nb) {
            int c = wn * 64 + nb * 16 + (lane & 15);   // 0..127 within half
            float bv = (EPI == 1) ? bias[colBlk + c] : 0.f;
            int rbase = rowBlk + wm * 32 + mb * 16 + ((lane & 16) ? 8 : 0);
#pragma unroll
            for (int v = 0; v < 8; ++v) {
                int r = rbase + v;
                float val = acc[mb][nb][v];
                if (EPI == 1) val += bv + addP[(size_t)r * 128 + c];
                outP[(size_t)r * ldOut + c] = val;
            }
        }
    }
}

// ---------------------------------------------------------------------------
// Kernel: in-place msg = (in_r + i*in_i) * (m_r + i*m_i)
// m layout: [N][256], cols 0-127 = real, 128-255 = imag
// ---------------------------------------------------------------------------
__global__ __launch_bounds__(256) void msg_kernel(
    float* __restrict__ m, const float* __restrict__ inR, const float* __restrict__ inI)
{
    int idx = blockIdx.x * 256 + threadIdx.x;   // 0 .. N*128-1
    int i = idx >> 7;
    int d = idx & 127;
    size_t base = (size_t)i * 256;
    float mr = m[base + d];
    float mi = m[base + 128 + d];
    float ir = inR[i], ii = inI[i];
    m[base + d]       = ir * mr - ii * mi;
    m[base + 128 + d] = ir * mi + ii * mr;
}

// ---------------------------------------------------------------------------
// Workspace layout (bytes)
// ---------------------------------------------------------------------------
#define OFF_ST    ((size_t)0)                         // bf16 [256][8192]  = 4 MB
#define OFF_PN    ((size_t)4194304)                   // f32  [N][10]      = 320 KB
#define OFF_INR   ((size_t)4521984)                   // f32  [N]
#define OFF_INI   ((size_t)4554752)                   // f32  [N]
#define OFF_M     ((size_t)4587520)                   // f32  [N][256]     = 8 MB
#define OFF_WBIG  ((size_t)12976128)                  // bf16 [256][256]   = 128 KB
#define OFF_BIAS2 ((size_t)13107200)                  // f32  [256]

extern "C" void kernel_launch(void* const* d_in, const int* in_sizes, int n_in,
                              void* d_out, int out_size, void* d_ws, size_t ws_size,
                              hipStream_t stream)
{
    const float* x_real     = (const float*)d_in[0];
    const float* x_imag     = (const float*)d_in[1];
    const float* adj        = (const float*)d_in[2];
    const float* theta_real = (const float*)d_in[3];
    const float* theta_imag = (const float*)d_in[4];
    const float* W_r        = (const float*)d_in[5];
    const float* b_r        = (const float*)d_in[6];
    const float* W_i        = (const float*)d_in[7];
    const float* b_i        = (const float*)d_in[8];
    const float* Wc_r       = (const float*)d_in[9];
    const float* bc_r       = (const float*)d_in[10];
    const float* Wc_i       = (const float*)d_in[11];
    const float* bc_i       = (const float*)d_in[12];
    const int*   labels     = (const int*)d_in[13];
    const int*   train_idx  = (const int*)d_in[14];

    char* w = (char*)d_ws;
    __bf16* St    = (__bf16*)(w + OFF_ST);
    float*  pn    = (float*)(w + OFF_PN);
    float*  inR   = (float*)(w + OFF_INR);
    float*  inI   = (float*)(w + OFF_INI);
    float*  mbuf  = (float*)(w + OFF_M);
    __bf16* Wbig  = (__bf16*)(w + OFF_WBIG);
    float*  bias2 = (float*)(w + OFF_BIAS2);

    float* out = (float*)d_out;
    float* outSep = out + (size_t)2 * NN * DD;
    float* outSup = outSep + 1;

    // 1) combined complex-as-real weight + bias
    wprep_kernel<<<256, 256, 0, stream>>>(W_r, W_i, b_r, b_i, Wbig, bias2);

    // 2) per-node: class proj, softmax, in_r/in_i, S^T (bf16)
    node_kernel<<<NN / 8, 256, 0, stream>>>(x_real, x_imag, Wc_r, bc_r, Wc_i, bc_i,
                                            theta_real, theta_imag, St, pn, inR, inI);

    // 3) losses
    loss_kernel<<<1, 256, 0, stream>>>(pn, labels, train_idx, theta_imag, outSep, outSup);

    // 4) big GEMM: m = adj @ [s_r | s_i]   (M=8192, K=8192, Ncols=256)
    gemm_wmma_kernel<0><<<dim3(NN / BM, 2), 256, 0, stream>>>(
        adj, NN, St, NN, NN,
        mbuf, mbuf + 128, 256,
        nullptr, nullptr, nullptr);

    // 5) msg = in * m (in place)
    msg_kernel<<<(NN * DD) / 256, 256, 0, stream>>>(mbuf, inR, inI);

    // 6) final complex MLP + residual: out = msg @ Wbig^T(combined) + bias2 + x
    gemm_wmma_kernel<1><<<dim3(NN / BM, 2), 256, 0, stream>>>(
        mbuf, 256, Wbig, 256, 256,
        out, out + (size_t)NN * DD, DD,
        bias2, x_real, x_imag);
}
